// Model_45930380263851
// MI455X (gfx1250) — compile-verified, matrix operand order checked by reading, not run on hardware
//
#include <hip/hip_runtime.h>
#include <hip/hip_bf16.h>
#include <math.h>

#define NS      32768
#define NFR     128
#define MD      128
#define NATOMS  16
#define NTF     384
#define NENVB   64
#define ENVF    128
#define NBANDS  128
#define KS      512
#define NL      6
#define NB      16
#define KPAD    288
#define LOG2NS  15

typedef __attribute__((ext_vector_type(16))) __bf16 v16bf;
typedef __attribute__((ext_vector_type(8)))  float  v8f;

union FragU { unsigned u[8]; v16bf v; };

__device__ inline v8f wmma_bf16(const unsigned* a, const unsigned* b, v8f c){
  FragU A, Bf;
#pragma unroll
  for (int i = 0; i < 8; ++i){ A.u[i] = a[i]; Bf.u[i] = b[i]; }
  return __builtin_amdgcn_wmma_f32_16x16x32_bf16(false, A.v, false, Bf.v,
                                                 (short)0, c, false, false);
}

__device__ inline unsigned short f2bf(float x){
  return (unsigned short)(__float_as_uint(x) >> 16);
}

// K offset of dword j within a 16-bit 16x32 A/B fragment (per CDNA5 VGPR layout):
// VGPR0..3 hold K={0,1..6,7}, VGPR4..7 hold K={16..23}; lanes 16-31 shift K by +8.
__device__ inline int frag_k(int j, int lane){
  return ((j < 4) ? (2 * j) : (8 + 2 * j)) + ((lane >> 4) & 1) * 8;
}

// Fragment builder from row-major f32 (truncating convert to bf16).
__device__ inline void build_frag(unsigned* fr, const float* src, int row0,
                                  int stride, int kbase, int lane){
  const float* p = src + (size_t)(row0 + (lane & 15)) * stride + kbase;
#pragma unroll
  for (int j = 0; j < 8; ++j){
    int k = frag_k(j, lane);
    unsigned lo = __float_as_uint(p[k]);
    unsigned hi = __float_as_uint(p[k + 1]);
    fr[j] = (hi & 0xFFFF0000u) | (lo >> 16);
  }
}

// Fragment builder from row-major bf16 (u16); strides/kbase even -> dword aligned b32.
__device__ inline void build_frag(unsigned* fr, const unsigned short* src, int row0,
                                  int stride, int kbase, int lane){
  const unsigned short* p = src + (size_t)(row0 + (lane & 15)) * stride + kbase;
#pragma unroll
  for (int j = 0; j < 8; ++j){
    int k = frag_k(j, lane);
    fr[j] = *(const unsigned*)(p + k);
  }
}

// D = A(MxK) * Bw(NxK)^T * scale + bias; optional activation / transposed store.
// A/B may each be f32 or pre-converted bf16 (overload-resolved fragment builders).
template<typename TA, typename TB>
__device__ inline void gemm_any(const TA* A, int lda, const TB* Bw, int ldb,
                                const float* bias, float* D, int ldd,
                                int M, int N, int K, float scale, int act, int transD,
                                int wave, int nwaves, int lane){
  int nm = M >> 4, nn = N >> 4, ntile = nm * nn, nk = K >> 5;
  for (int tix = wave; tix < ntile; tix += nwaves){
    int m0 = (tix % nm) << 4;
    int n0 = (tix / nm) << 4;
    v8f acc = {0.f,0.f,0.f,0.f,0.f,0.f,0.f,0.f};
    for (int kt = 0; kt < nk; ++kt){
      unsigned af[8], bf[8];
      build_frag(af, A,  m0, lda, kt << 5, lane);
      build_frag(bf, Bw, n0, ldb, kt << 5, lane);
      acc = wmma_bf16(af, bf, acc);
    }
    int col = n0 + (lane & 15);
    int mo  = ((lane >> 4) & 1) * 8;
    float bv = bias ? bias[col] : 0.0f;
#pragma unroll
    for (int r = 0; r < 8; ++r){
      float v = acc[r] * scale + bv;
      if (act == 1)      v = 0.5f * v * (1.0f + erff(v * 0.70710678f));   // gelu
      else if (act == 2) v = (v > 0.0f) ? v : 0.2f * v;                   // leaky
      int row = m0 + r + mo;
      if (transD) D[(size_t)col * ldd + row] = v;
      else        D[(size_t)row * ldd + col] = v;
    }
  }
}

// ---------------- prep kernels ----------------
__global__ void k_f2bf_arr(const float* src, unsigned short* dst, int n){
  int i = blockIdx.x * blockDim.x + threadIdx.x;
  if (i < n) dst[i] = f2bf(src[i]);
}

__global__ void k_prep_tf(const float* re, const float* im,
                          unsigned short* reb, unsigned short* imb){
  int i = blockIdx.x * blockDim.x + threadIdx.x;
  if (i >= NTF * NFR * KPAD) return;
  int k  = i % KPAD;
  int rf = i / KPAD;
  float vr = 0.f, vi = 0.f;
  if (k < 257){ vr = re[(size_t)rf * 257 + k]; vi = im[(size_t)rf * 257 + k]; }
  reb[i] = f2bf(vr); imb[i] = f2bf(vi);
}

__global__ void k_prep_basis(unsigned short* C, unsigned short* S){
  int i = blockIdx.x * blockDim.x + threadIdx.x;
  if (i >= KS * KPAD) return;
  int k = i % KPAD, t = i / KPAD;
  float c = 0.f, s = 0.f;
  if (k < 257){
    float ck = (k == 0 || k == 256) ? 1.0f : 2.0f;
    float ang = 6.283185307179586f * (float)k * (float)t / 512.0f;
    float sv, cv; __sincosf(ang, &sv, &cv);
    float nrm = ck * 0.044194173824159216f;   // 1/sqrt(512), ortho irfft
    c = cv * nrm; s = -sv * nrm;
  }
  C[i] = f2bf(c); S[i] = f2bf(s);
}

__global__ void k_prep_env(const float* env, float* envn){
  __shared__ float red[128];
  int b = blockIdx.x, t = threadIdx.x;
  float v = env[b * ENVF + t];
  red[t] = v; __syncthreads();
  for (int s = 64; s > 0; s >>= 1){
    if (t < s) red[t] = fmaxf(red[t], red[t + s]);
    __syncthreads();
  }
  envn[b * ENVF + t] = v / (red[0] + 1e-8f);
}

// ---------------- fused filterbank conv + |.| + avg-pool (WMMA bf16) ----------------
// One block = (batch, 256-sample strip). M=time, N=filter, K=512 taps.
// launch_bounds(256,1): filter B-fragments are reloaded from global for each (mt, kt).
__global__ __launch_bounds__(256, 1)
void k_conv_pool(const float* x, const unsigned short* filt, float* pooled){
  __shared__ unsigned short xw[1024];
  __shared__ float pools[NBANDS];
  int tid = threadIdx.x;
  int blk = blockIdx.x;
  int b  = blk >> 7;
  int n0 = blk & 127;
  int t0 = n0 << 8;
  const float* xb = x + (size_t)b * NS;
  __builtin_prefetch(filt + tid * 256, 0, 3);       // pull filter bank toward L2
  for (int i = tid; i < 1024; i += 256){
    int g = t0 - 255 + i;
    float v = (i < 768 && g >= 0 && g < NS) ? xb[g] : 0.0f;
    xw[i] = f2bf(v);
  }
  if (tid < NBANDS) pools[tid] = 0.0f;
  __syncthreads();

  int wave = tid >> 5, lane = tid & 31;
  int f0 = wave << 4;                               // 8 waves cover 128 filters
  for (int mt = 0; mt < 16; ++mt){
    v8f acc = {0.f,0.f,0.f,0.f,0.f,0.f,0.f,0.f};
    int m = (mt << 4) + (lane & 15);
    for (int kt = 0; kt < 16; ++kt){
      int kb = kt << 5;
      unsigned af[8], bf[8];
#pragma unroll
      for (int j = 0; j < 8; ++j){
        int k = kb + frag_k(j, lane);
        unsigned lo = xw[m + k];
        unsigned hi = xw[m + k + 1];
        af[j] = lo | (hi << 16u);
      }
      build_frag(bf, filt, f0, KS, kb, lane);
      acc = wmma_bf16(af, bf, acc);
    }
    float cs = 0.0f;
#pragma unroll
    for (int r = 0; r < 8; ++r) cs += fabsf(acc[r]);
    cs += __shfl_xor(cs, 16, 32);                   // fold M halves
    if (lane < 16) pools[f0 + lane] += cs;          // per-wave exclusive filters
  }
  __syncthreads();
  if (tid < NBANDS){
    float v = pools[tid] * (1.0f / 512.0f);
    atomicAdd(&pooled[((size_t)b * NBANDS + tid) * NFR + n0], v);
    if (n0 + 1 < NFR)
      atomicAdd(&pooled[((size_t)b * NBANDS + tid) * NFR + n0 + 1], v);
  }
}

// ---------------- positional-encoding + embedding ----------------
__global__ void k_embed(const float* pooled, const float* ew, const float* eb, float* H){
  int blk = blockIdx.x;
  int b = blk >> 7, t = blk & 127;
  int o = threadIdx.x;
  float pos = -3.14159265358979f + (float)t * (6.28318530717959f / 127.0f);
  const float* wr = ew + (size_t)o * 161;
  float acc = eb[o] + wr[0] * pos;
  float p2 = pos;
#pragma unroll
  for (int i = 0; i < 16; ++i){
    float sv, cv; __sincosf(p2, &sv, &cv);
    acc += wr[1 + 2 * i] * sv + wr[2 + 2 * i] * cv;
    p2 *= 2.0f;
  }
  const float* pb = pooled + (size_t)b * NBANDS * NFR;
  for (int f = 0; f < NBANDS; ++f) acc += wr[33 + f] * pb[f * NFR + t];
  H[((size_t)b * NFR + t) * MD + o] = acc;
}

// ---------------- full transformer encoder + top-k atom select ----------------
// LDS: hb (f32 residual, 64KB) + scr (f32 scratch, 64KB) + hbf (bf16 shadow, 32KB).
__global__ __launch_bounds__(256, 1)
void k_encoder(const float* Hin,
      const unsigned short* wqkv, const float* bqkv,
      const unsigned short* wo,   const float* bo,
      const float* ln1g, const float* ln1b, const float* ln2g, const float* ln2b,
      const unsigned short* w1, const float* b1,
      const unsigned short* w2, const float* b2,
      const float* attw, const float* attb,
      float* qkvG, float* vtG, float* oG, float* Z,
      float* vecOut, int* idxOut){
  extern __shared__ float dyn[];
  float* hb  = dyn;                                  // 128x128 residual stream
  float* scr = dyn + 16384;                          // 128x128 scratch
  unsigned short* hbf = (unsigned short*)(dyn + 32768);  // bf16 shadow of hb
  __shared__ float lg[NFR];
  __shared__ int sel[NATOMS];

  int b = blockIdx.x, tid = threadIdx.x;
  int wave = tid >> 5, lane = tid & 31;

  const float* Hb = Hin + (size_t)b * NFR * MD;
  for (int i = tid; i < NFR * MD; i += 256){
    float v = Hb[i];
    hb[i] = v; hbf[i] = f2bf(v);
  }
  float* qB  = qkvG + (size_t)b * 3 * NFR * MD;
  float* kB  = qB + NFR * MD;
  float* vtB = vtG + (size_t)b * NFR * MD;
  float* oB  = oG  + (size_t)b * NFR * MD;
  __syncthreads();

  for (int l = 0; l < NL; ++l){
    const unsigned short* wl = wqkv + (size_t)l * 3 * MD * MD;
    const float* bl = bqkv + (size_t)l * 3 * MD;
    gemm_any(hbf, MD, wl,           MD, bl,        qB,  MD, NFR, MD, MD, 1.f, 0, 0, wave, 8, lane);
    gemm_any(hbf, MD, wl + MD*MD,   MD, bl + MD,   kB,  MD, NFR, MD, MD, 1.f, 0, 0, wave, 8, lane);
    gemm_any(hbf, MD, wl + 2*MD*MD, MD, bl + 2*MD, vtB, MD, NFR, MD, MD, 1.f, 0, 1, wave, 8, lane); // V^T
    __syncthreads();
    for (int hd = 0; hd < 4; ++hd){
      gemm_any(qB + hd*32, MD, kB + hd*32, MD, (const float*)nullptr, scr, NFR,
               NFR, NFR, 32, 0.17677669529663687f, 0, 0, wave, 8, lane);
      __syncthreads();
      if (tid < NFR){
        float* row = scr + tid * NFR;
        float mx = -1e30f;
        for (int j = 0; j < NFR; ++j) mx = fmaxf(mx, row[j]);
        float s = 0.f;
        for (int j = 0; j < NFR; ++j){ float e = __expf(row[j] - mx); row[j] = e; s += e; }
        float inv = 1.0f / s;
        for (int j = 0; j < NFR; ++j) row[j] *= inv;
      }
      __syncthreads();
      gemm_any(scr, NFR, vtB + hd*32*MD, MD, (const float*)nullptr, oB + hd*32, MD,
               NFR, 32, NFR, 1.f, 0, 0, wave, 8, lane);
      __syncthreads();
    }
    gemm_any(oB, MD, wo + (size_t)l*MD*MD, MD, bo + l*MD, scr, MD,
             NFR, MD, MD, 1.f, 0, 0, wave, 8, lane);
    __syncthreads();
    if (tid < NFR){                         // residual + LN1 (+ refresh bf16 shadow)
      const float* g = ln1g + l*MD; const float* be = ln1b + l*MD;
      float* hr = hb + tid*MD; float* sr = scr + tid*MD;
      float m = 0.f;
      for (int c = 0; c < MD; ++c) m += hr[c] + sr[c];
      m *= (1.0f / MD);
      float v = 0.f;
      for (int c = 0; c < MD; ++c){ float d = hr[c] + sr[c] - m; v += d * d; }
      float inv = rsqrtf(v * (1.0f / MD) + 1e-5f);
      for (int c = 0; c < MD; ++c){
        float nv = (hr[c] + sr[c] - m) * inv * g[c] + be[c];
        hr[c] = nv; hbf[tid*MD + c] = f2bf(nv);
      }
    }
    __syncthreads();
    gemm_any(hbf, MD, w1 + (size_t)l*MD*MD, MD, b1 + l*MD, qB, MD,
             NFR, MD, MD, 1.f, 1, 0, wave, 8, lane);        // gelu
    __syncthreads();
    gemm_any(qB, MD, w2 + (size_t)l*MD*MD, MD, b2 + l*MD, scr, MD,
             NFR, MD, MD, 1.f, 0, 0, wave, 8, lane);
    __syncthreads();
    if (tid < NFR){                         // residual + LN2 (+ refresh bf16 shadow)
      const float* g = ln2g + l*MD; const float* be = ln2b + l*MD;
      float* hr = hb + tid*MD; float* sr = scr + tid*MD;
      float m = 0.f;
      for (int c = 0; c < MD; ++c) m += hr[c] + sr[c];
      m *= (1.0f / MD);
      float v = 0.f;
      for (int c = 0; c < MD; ++c){ float d = hr[c] + sr[c] - m; v += d * d; }
      float inv = rsqrtf(v * (1.0f / MD) + 1e-5f);
      for (int c = 0; c < MD; ++c){
        float nv = (hr[c] + sr[c] - m) * inv * g[c] + be[c];
        hr[c] = nv; hbf[tid*MD + c] = f2bf(nv);
      }
    }
    __syncthreads();
  }
  // unit-normalize rows, emit vectors output + attention logits
  if (tid < NFR){
    float* hr = hb + tid * MD;
    float s = 0.f;
    for (int c = 0; c < MD; ++c) s += hr[c] * hr[c];
    float inv = 1.0f / (sqrtf(s) + 1e-12f);
    float lgv = attb[0];
    for (int c = 0; c < MD; ++c){
      float hn = hr[c] * inv;
      scr[tid * MD + c] = hn;
      vecOut[(size_t)b * MD * NFR + c * NFR + tid] = hn;
      lgv += attw[c] * hn;
    }
    lg[tid] = lgv;
  }
  __syncthreads();
  if (tid == 0){
    for (int j = 0; j < NATOMS; ++j){
      int bi = 0; float bv = -1e30f;
      for (int t = 0; t < NFR; ++t) if (lg[t] > bv){ bv = lg[t]; bi = t; }
      sel[j] = bi; lg[bi] = -1e30f;
      idxOut[b * NATOMS + j] = bi;
    }
  }
  __syncthreads();
  for (int j = 0; j < NATOMS; ++j)
    for (int c = tid; c < MD; c += 256)
      Z[((size_t)b * NATOMS + j) * MD + c] = scr[sel[j] * MD + c];
}

// ---------------- MLP heads (tf / env / amp) ----------------
__device__ inline void mlp3(const float* w, const float* bias, float* zin, int tid){
  for (int li = 0; li < 3; ++li){
    float acc = bias[li * MD + tid];
    const float* wr = w + (size_t)(li * MD + tid) * MD;
    for (int k = 0; k < MD; ++k) acc += zin[k] * wr[k];
    acc = (acc > 0.f) ? acc : 0.2f * acc;
    __syncthreads();
    zin[tid] = acc;
    __syncthreads();
  }
}

__global__ void k_heads(const float* Z,
   const float* tsw, const float* tsb, const float* tow, const float* tob,
   const float* esw, const float* esb, const float* eow, const float* eob,
   const float* asw, const float* asb, const float* aow, const float* aob,
   int* TI, int* EI, float* AMP){
  __shared__ float zin[MD];
  __shared__ float lg[NTF];
  int atom = blockIdx.x, tid = threadIdx.x;
  const float* zr = Z + (size_t)atom * MD;

  zin[tid] = zr[tid]; __syncthreads();
  mlp3(tsw, tsb, zin, tid);
  for (int o = tid; o < NTF; o += MD){
    float acc = tob[o];
    const float* wr = tow + (size_t)o * MD;
    for (int k = 0; k < MD; ++k) acc += zin[k] * wr[k];
    lg[o] = acc;
  }
  __syncthreads();
  if (tid == 0){
    int bi = 0; float bv = -1e30f;
    for (int o = 0; o < NTF; ++o) if (lg[o] > bv){ bv = lg[o]; bi = o; }
    TI[atom] = bi;
  }
  __syncthreads();

  zin[tid] = zr[tid]; __syncthreads();
  mlp3(esw, esb, zin, tid);
  if (tid < NENVB){
    float acc = eob[tid];
    const float* wr = eow + (size_t)tid * MD;
    for (int k = 0; k < MD; ++k) acc += zin[k] * wr[k];
    lg[tid] = acc;
  }
  __syncthreads();
  if (tid == 0){
    int bi = 0; float bv = -1e30f;
    for (int o = 0; o < NENVB; ++o) if (lg[o] > bv){ bv = lg[o]; bi = o; }
    EI[atom] = bi;
  }
  __syncthreads();

  zin[tid] = zr[tid]; __syncthreads();
  mlp3(asw, asb, zin, tid);
  if (tid == 0){
    float acc = aob[0];
    for (int k = 0; k < MD; ++k) acc += zin[k] * aow[k];
    AMP[atom] = fabsf(acc);
  }
}

// ---------------- selected-row iDFT (WMMA) + fused overlap-add + normalize ----------------
__global__ __launch_bounds__(256, 1)
void k_funcs(const unsigned short* reb, const unsigned short* imb,
             const unsigned short* basC, const unsigned short* basS,
             const int* TI, float* funcs){
  extern __shared__ float acb[];            // 33024 accum + 256 reduce
  float* red = acb + 33024;
  int atom = blockIdx.x, tid = threadIdx.x;
  int wave = tid >> 5, lane = tid & 31;
  for (int i = tid; i < 33024; i += 256) acb[i] = 0.f;
  __syncthreads();
  int nf = TI[atom];
  const unsigned short* ar = reb + (size_t)nf * NFR * KPAD;
  const unsigned short* ai = imb + (size_t)nf * NFR * KPAD;
  __builtin_prefetch(ar + tid * 144, 0, 1);
  for (int tix = wave; tix < 256; tix += 8){       // 8 M-tiles x 32 N-tiles
    int fr0 = (tix & 7) << 4;
    int t0  = (tix >> 3) << 4;
    v8f acc = {0.f,0.f,0.f,0.f,0.f,0.f,0.f,0.f};
    for (int kt = 0; kt < 9; ++kt){                // K = 288 (257 padded)
      int kb = kt << 5;
      unsigned af[8], bf[8];
      build_frag(af, ar,   fr0, KPAD, kb, lane);
      build_frag(bf, basC, t0,  KPAD, kb, lane);
      acc = wmma_bf16(af, bf, acc);
      build_frag(af, ai,   fr0, KPAD, kb, lane);
      build_frag(bf, basS, t0,  KPAD, kb, lane);
      acc = wmma_bf16(af, bf, acc);
    }
    int t  = t0 + (lane & 15);
    int mo = ((lane >> 4) & 1) * 8;
#pragma unroll
    for (int r = 0; r < 8; ++r){
      int fr = fr0 + r + mo;
      atomicAdd(&acb[fr * 256 + t], acc[r]);       // overlap-add: p = fr*256 + t
    }
  }
  __syncthreads();
  float mx = -1e30f;
  for (int i = tid; i < NS; i += 256) mx = fmaxf(mx, acb[i]);
  red[tid] = mx; __syncthreads();
  for (int s = 128; s > 0; s >>= 1){
    if (tid < s) red[tid] = fmaxf(red[tid], red[tid + s]);
    __syncthreads();
  }
  float inv = 1.0f / (red[0] + 1e-8f);
  float* out = funcs + (size_t)atom * NS;
  for (int i = tid; i < NS; i += 256) out[i] = acb[i] * inv;
}

// ---------------- 32768-pt radix-2 FFT fully in LDS (256 KB of the 320 KB WGP pool) ----
__device__ void fft_sh(float2* d, int tid, int nt, float sign){
  for (int i = tid; i < NS; i += nt){
    unsigned j = __brev((unsigned)i) >> (32 - LOG2NS);
    if (j > (unsigned)i){ float2 t = d[i]; d[i] = d[j]; d[j] = t; }
  }
  __syncthreads();
  for (int s = 1; s <= LOG2NS; ++s){
    int m = 1 << s, half = m >> 1;
    for (int k = tid; k < (NS >> 1); k += nt){
      int g = k / half, j = k - g * half;
      int i0 = g * m + j, i1 = i0 + half;
      float ang = sign * 6.283185307179586f * (float)j / (float)m;
      float sv, cv; __sincosf(ang, &sv, &cv);
      float2 u = d[i0], t = d[i1];
      float tr = t.x * cv - t.y * sv;
      float ti = t.x * sv + t.y * cv;
      d[i0] = make_float2(u.x + tr, u.y + ti);
      d[i1] = make_float2(u.x - tr, u.y - ti);
    }
    __syncthreads();
  }
}

__global__ void k_fftconv(const float* funcs, const float* envn, const float* noise,
                          const int* EI, const float* AMP, float2* fspec, float* atoms){
  extern __shared__ float2 sm[];
  int atom = blockIdx.x, tid = threadIdx.x;
  const float* fr = funcs + (size_t)atom * NS;
  const float* nr = noise + (size_t)atom * NS;
  for (int i = tid; i < NS; i += 1024){
    __builtin_prefetch(nr + i, 0, 1);               // stream noise toward caches
    sm[i] = make_float2(fr[i], 0.f);
  }
  __syncthreads();
  fft_sh(sm, tid, 1024, -1.0f);
  float2* fs = fspec + (size_t)atom * NS;
  for (int i = tid; i < NS; i += 1024) fs[i] = sm[i];
  __syncthreads();
  const float* er = envn + EI[atom] * ENVF;
  for (int i = tid; i < NS; i += 1024){
    float c = ((float)i + 0.5f) * (128.0f / 32768.0f) - 0.5f;
    c = fminf(fmaxf(c, 0.0f), 127.0f);
    int lo = (int)c;
    int hi = min(lo + 1, 127);
    float w = c - (float)lo;
    float ev = fabsf(er[lo]) * (1.0f - w) + fabsf(er[hi]) * w;
    sm[i] = make_float2(ev * (nr[i] * 2.0f - 1.0f), 0.f);
  }
  __syncthreads();
  fft_sh(sm, tid, 1024, -1.0f);
  for (int i = tid; i < NS; i += 1024){
    float2 a = sm[i], bv = fs[i];
    sm[i] = make_float2(a.x * bv.x - a.y * bv.y, a.x * bv.y + a.y * bv.x);
  }
  __syncthreads();
  fft_sh(sm, tid, 1024, 1.0f);
  float scale = AMP[atom] * (1.0f / (32768.0f * 181.01933598375618f)); // ortho net 1/sqrt(N)
  float* ot = atoms + (size_t)atom * NS;
  for (int i = tid; i < NS; i += 1024) ot[i] = sm[i].x * scale;
}

// ---------------- scatter-add and final normalize ----------------
__global__ void k_scatter(const float* atoms, const int* idx, float* outacc){
  int atom = blockIdx.x, tid = threadIdx.x;
  int b = atom >> 4, j = atom & 15;
  int start = idx[b * NATOMS + j] << 8;
  const float* ar = atoms + (size_t)atom * NS;
  float* ob = outacc + (size_t)b * 65536 + start;
  for (int i = tid; i < NS; i += 256) atomicAdd(&ob[i], ar[i]);
}

__global__ void k_final(const float* outacc, float* out){
  __shared__ float red[256];
  int b = blockIdx.x, tid = threadIdx.x;
  const float* ob = outacc + (size_t)b * 65536;
  float mx = 0.f;
  for (int i = tid; i < NS; i += 256) mx = fmaxf(mx, fabsf(ob[i]));
  red[tid] = mx; __syncthreads();
  for (int s = 128; s > 0; s >>= 1){
    if (tid < s) red[tid] = fmaxf(red[tid], red[tid + s]);
    __syncthreads();
  }
  float inv = 1.0f / (red[0] + 1e-12f);
  float* o = out + (size_t)b * NS;
  for (int i = tid; i < NS; i += 256) o[i] = ob[i] * inv;
}

// ---------------- launcher ----------------
extern "C" void kernel_launch(void* const* d_in, const int* in_sizes, int n_in,
                              void* d_out, int out_size, void* d_ws, size_t ws_size,
                              hipStream_t stream){
  (void)in_sizes; (void)n_in; (void)out_size; (void)ws_size;
  const float* x        = (const float*)d_in[0];
  const float* fb       = (const float*)d_in[1];
  const float* embed_w  = (const float*)d_in[2];
  const float* embed_b  = (const float*)d_in[3];
  const float* enc_wqkv = (const float*)d_in[4];
  const float* enc_bqkv = (const float*)d_in[5];
  const float* enc_wo   = (const float*)d_in[6];
  const float* enc_bo   = (const float*)d_in[7];
  const float* ln1g     = (const float*)d_in[8];
  const float* ln1b     = (const float*)d_in[9];
  const float* ln2g     = (const float*)d_in[10];
  const float* ln2b     = (const float*)d_in[11];
  const float* enc_w1   = (const float*)d_in[12];
  const float* enc_b1   = (const float*)d_in[13];
  const float* enc_w2   = (const float*)d_in[14];
  const float* enc_b2   = (const float*)d_in[15];
  const float* attw     = (const float*)d_in[16];
  const float* attb     = (const float*)d_in[17];
  const float* tf_real  = (const float*)d_in[18];
  const float* tf_imag  = (const float*)d_in[19];
  const float* envel    = (const float*)d_in[20];
  const float* tsw      = (const float*)d_in[21];
  const float* tsb      = (const float*)d_in[22];
  const float* tow      = (const float*)d_in[23];
  const float* tob      = (const float*)d_in[24];
  const float* esw      = (const float*)d_in[25];
  const float* esb      = (const float*)d_in[26];
  const float* eow      = (const float*)d_in[27];
  const float* eob      = (const float*)d_in[28];
  const float* asw      = (const float*)d_in[29];
  const float* asb      = (const float*)d_in[30];
  const float* aow      = (const float*)d_in[31];
  const float* aob      = (const float*)d_in[32];
  const float* noise    = (const float*)d_in[33];

  char* wsb = (char*)d_ws;
  size_t off = 0;
  auto alloc = [&](size_t bytes)->char*{
    char* p = wsb + off; off += (bytes + 255) & ~(size_t)255; return p;
  };
  unsigned short* filt_bf = (unsigned short*)alloc((size_t)NBANDS * KS * 2);
  unsigned short* tfre_bf = (unsigned short*)alloc((size_t)NTF * NFR * KPAD * 2);
  unsigned short* tfim_bf = (unsigned short*)alloc((size_t)NTF * NFR * KPAD * 2);
  unsigned short* basC    = (unsigned short*)alloc((size_t)KS * KPAD * 2);
  unsigned short* basS    = (unsigned short*)alloc((size_t)KS * KPAD * 2);
  unsigned short* wqkv_bf = (unsigned short*)alloc((size_t)NL * 3 * MD * MD * 2);
  unsigned short* wo_bf   = (unsigned short*)alloc((size_t)NL * MD * MD * 2);
  unsigned short* w1_bf   = (unsigned short*)alloc((size_t)NL * MD * MD * 2);
  unsigned short* w2_bf   = (unsigned short*)alloc((size_t)NL * MD * MD * 2);
  float*  envn   = (float*)alloc((size_t)NENVB * ENVF * 4);
  float*  pooled = (float*)alloc((size_t)NB * NBANDS * NFR * 4);
  float*  H      = (float*)alloc((size_t)NB * NFR * MD * 4);
  float*  qkvG   = (float*)alloc((size_t)NB * 3 * NFR * MD * 4);
  float*  vtG    = (float*)alloc((size_t)NB * NFR * MD * 4);
  float*  oG     = (float*)alloc((size_t)NB * NFR * MD * 4);
  float*  Z      = (float*)alloc((size_t)256 * MD * 4);
  int*    TI     = (int*)alloc(256 * 4);
  int*    EI     = (int*)alloc(256 * 4);
  float*  AMP    = (float*)alloc(256 * 4);
  float*  funcs  = (float*)alloc((size_t)256 * NS * 4);
  float2* fspec  = (float2*)alloc((size_t)256 * NS * 8);
  float*  atoms  = (float*)alloc((size_t)256 * NS * 4);
  float*  outacc = (float*)alloc((size_t)NB * 65536 * 4);

  float* out_region = (float*)d_out;
  int*   idx_region = (int*)((float*)d_out + (size_t)NB * NS);
  float* vec_region = (float*)d_out + (size_t)NB * NS + NB * NATOMS;

  hipMemsetAsync(pooled, 0, (size_t)NB * NBANDS * NFR * 4, stream);
  hipMemsetAsync(outacc, 0, (size_t)NB * 65536 * 4, stream);

  k_f2bf_arr<<<(NBANDS * KS + 255) / 256, 256, 0, stream>>>(fb, filt_bf, NBANDS * KS);
  k_f2bf_arr<<<(NL*3*MD*MD + 255) / 256, 256, 0, stream>>>(enc_wqkv, wqkv_bf, NL*3*MD*MD);
  k_f2bf_arr<<<(NL*MD*MD + 255) / 256, 256, 0, stream>>>(enc_wo, wo_bf, NL*MD*MD);
  k_f2bf_arr<<<(NL*MD*MD + 255) / 256, 256, 0, stream>>>(enc_w1, w1_bf, NL*MD*MD);
  k_f2bf_arr<<<(NL*MD*MD + 255) / 256, 256, 0, stream>>>(enc_w2, w2_bf, NL*MD*MD);
  {
    int tot = NTF * NFR * KPAD;
    k_prep_tf<<<(tot + 255) / 256, 256, 0, stream>>>(tf_real, tf_imag, tfre_bf, tfim_bf);
  }
  k_prep_basis<<<(KS * KPAD + 255) / 256, 256, 0, stream>>>(basC, basS);
  k_prep_env<<<NENVB, ENVF, 0, stream>>>(envel, envn);

  k_conv_pool<<<NB * NFR, 256, 0, stream>>>(x, filt_bf, pooled);
  k_embed<<<NB * NFR, MD, 0, stream>>>(pooled, embed_w, embed_b, H);
  k_encoder<<<NB, 256, (16384 + 16384 + 8192) * sizeof(float), stream>>>(H,
        wqkv_bf, enc_bqkv, wo_bf, enc_bo, ln1g, ln1b, ln2g, ln2b,
        w1_bf, enc_b1, w2_bf, enc_b2, attw, attb,
        qkvG, vtG, oG, Z, vec_region, idx_region);
  k_heads<<<256, MD, 0, stream>>>(Z, tsw, tsb, tow, tob, esw, esb, eow, eob,
                                  asw, asb, aow, aob, TI, EI, AMP);
  k_funcs<<<256, 256, (33024 + 256) * sizeof(float), stream>>>(
        tfre_bf, tfim_bf, basC, basS, TI, funcs);
  k_fftconv<<<256, 1024, NS * sizeof(float2), stream>>>(
        funcs, envn, noise, EI, AMP, fspec, atoms);
  k_scatter<<<256, 256, 0, stream>>>(atoms, idx_region, outacc);
  k_final<<<NB, 256, 0, stream>>>(outacc, out_region);
}
